// LlamaForCausalLMWarpper_2808908612221
// MI455X (gfx1250) — matrix operand order repaired and checked by reading, not compiled, not run on hardware
//
#include <hip/hip_runtime.h>
#include <hip/hip_bf16.h>
#include <math.h>

// ---------------------------------------------------------------------------
// Fused linear + cross-entropy (LM head loss) for MI455X / gfx1250.
//   hidden [1,2048,4096] f32, weight [32000,4096] f32, labels [1,2048] i32
//   loss = mean over shifted tokens of -log_softmax(hs @ W^T)[label]
//
// Strategy:
//   * bf16 V_WMMA_F32_16X16X32 GEMM, f32 accumulate (reference casts to f32).
//   * A tile (32 tokens x 4096) staged once per block in 256 KB of LDS (bf16).
//   * W preconverted f32->bf16 into workspace when ws_size allows: halves the
//     streamed W bytes and removes all cvts from the hot loop. Deterministic
//     branch on ws_size; convert re-runs every call (no cached state).
//   * Each wave processes TWO 16-col vocab tiles per K sweep: 4 WMMAs per
//     32-K step against 4 ds_b128 + 4 global_b128 loads.
//   * Single-pass online log-sum-exp per token, label logit captured inline,
//     lane butterfly + LDS wave combine + split-vocab partials in workspace.
// ---------------------------------------------------------------------------

typedef __bf16 bf16_t;
typedef __attribute__((ext_vector_type(8)))  __bf16 v8bf;
typedef __attribute__((ext_vector_type(16))) __bf16 v16bf;
typedef __attribute__((ext_vector_type(8)))  float  v8f;

#define HDIM   4096
#define VOCAB  32000
#define QLEN   2048
#define NTOK   (QLEN - 1)          // 2047 shifted tokens
#define MTILE  32                  // tokens per workgroup (A tile rows)
#define NWAVES 8
#define THREADS (NWAVES * 32)
#define VSPLIT 4                   // vocab splits across grid.y
#define NGROUPS (VOCAB / 32)       // 1000 groups of 2 vocab tiles (32 cols)
#define IGNORE_INDEX (-100)
#define L2E 1.44269504088896340736f
#define LN2 0.69314718055994530942f
#define A_BYTES ((size_t)MTILE * HDIM * 2)   // 256 KB bf16 A tile in LDS

__device__ __forceinline__ float fexp2(float x) { return __builtin_amdgcn_exp2f(x); }
__device__ __forceinline__ float flog2(float x) { return __builtin_amdgcn_logf(x); }

// Load a 16-bit A/B fragment (16 bf16) for this lane from a bf16 row pointer.
__device__ __forceinline__ v16bf load_frag_bf16(const bf16_t* p)
{
    const v8bf* q = (const v8bf*)p;
    return __builtin_shufflevector(q[0], q[2],
        0,1,2,3,4,5,6,7,8,9,10,11,12,13,14,15);
}

// Build a B fragment from f32 [0..7] and [16..23] (fallback path, in-loop convert).
__device__ __forceinline__ v16bf load_frag_f32(const float* p)
{
    const float4* wp = (const float4*)p;
    float4 w0 = wp[0], w1 = wp[1], w2 = wp[4], w3 = wp[5];
    v16bf b;
    b[0]=(bf16_t)w0.x;  b[1]=(bf16_t)w0.y;  b[2]=(bf16_t)w0.z;  b[3]=(bf16_t)w0.w;
    b[4]=(bf16_t)w1.x;  b[5]=(bf16_t)w1.y;  b[6]=(bf16_t)w1.z;  b[7]=(bf16_t)w1.w;
    b[8]=(bf16_t)w2.x;  b[9]=(bf16_t)w2.y;  b[10]=(bf16_t)w2.z; b[11]=(bf16_t)w2.w;
    b[12]=(bf16_t)w3.x; b[13]=(bf16_t)w3.y; b[14]=(bf16_t)w3.z; b[15]=(bf16_t)w3.w;
    return b;
}

template<bool PRECONV>
__global__ __launch_bounds__(THREADS)
void fused_ce_kernel(const float*  __restrict__ hs,      // [QLEN][HDIM]
                     const int*    __restrict__ labels,  // [QLEN]
                     const float*  __restrict__ Wf,      // [VOCAB][HDIM] f32
                     const bf16_t* __restrict__ Wb,      // [VOCAB][HDIM] bf16
                     float*        __restrict__ part)    // [NTOK][VSPLIT][3]
{
    extern __shared__ char smem[];
    bf16_t* As  = (bf16_t*)smem;                    // [MTILE][HDIM] bf16
    float*  red = (float*)(smem + A_BYTES);         // [NWAVES][MTILE][3]

    const int tid   = threadIdx.x;
    const int wave  = tid >> 5;
    const int lane  = tid & 31;
    const int col   = lane & 15;       // column within 16-wide vocab tile
    const int khalf = lane >> 4;       // which K half this lane holds
    const int rowbase = blockIdx.x * MTILE;
    const int vs      = blockIdx.y;

    // ---- Stage A tile: f32 -> bf16 into LDS (zeros for out-of-range rows) ----
    for (int idx = tid * 4; idx < MTILE * HDIM; idx += THREADS * 4) {
        int r = idx >> 12;             // / HDIM
        int k = idx & (HDIM - 1);
        int g = rowbase + r;
        float4 v = make_float4(0.f, 0.f, 0.f, 0.f);
        if (g < NTOK) v = *(const float4*)(hs + (size_t)g * HDIM + k);
        bf16_t* dst = As + (size_t)r * HDIM + k;
        dst[0] = (bf16_t)v.x; dst[1] = (bf16_t)v.y;
        dst[2] = (bf16_t)v.z; dst[3] = (bf16_t)v.w;
    }
    __syncthreads();

    // ---- per-lane labels for the 16 row slots this lane's C elements map to
    int lab0[8], lab1[8];
#pragma unroll
    for (int r = 0; r < 8; ++r) {
        int g0 = rowbase + khalf * 8 + r;     // rows 0..15 half
        int g1 = g0 + 16;                     // rows 16..31 half
        lab0[r] = (g0 < NTOK) ? labels[g0 + 1] : -2;
        lab1[r] = (g1 < NTOK) ? labels[g1 + 1] : -2;
    }

    // ---- online softmax running state (per lane: 8 rows x my column subset)
    float rm0[8], rs0[8], ll0[8], rm1[8], rs1[8], ll1[8];
#pragma unroll
    for (int r = 0; r < 8; ++r) {
        rm0[r] = rm1[r] = -INFINITY;
        rs0[r] = rs1[r] = 0.f;
        ll0[r] = ll1[r] = -INFINITY;
    }

    const bf16_t* a0base = As + (size_t)(lane & 15) * HDIM + khalf * 8;
    const bf16_t* a1base = a0base + (size_t)16 * HDIM;

    // ---- sweep this (wave, vsplit)'s share of 32-column vocab groups ----
    for (int g = vs * NWAVES + wave; g < NGROUPS; g += VSPLIT * NWAVES) {
        const int vbase = g * 32;
        const int vcol0 = vbase + col;        // tile 0 column
        const int vcol1 = vcol0 + 16;         // tile 1 column

        const bf16_t* wb0 = Wb + (size_t)vcol0 * HDIM + khalf * 8;
        const bf16_t* wb1 = Wb + (size_t)vcol1 * HDIM + khalf * 8;
        const float*  wf0 = Wf + (size_t)vcol0 * HDIM + khalf * 8;
        const float*  wf1 = Wf + (size_t)vcol1 * HDIM + khalf * 8;

        v8f acc00 = {};   // rows 0..15  x tile 0
        v8f acc01 = {};   // rows 16..31 x tile 0
        v8f acc10 = {};   // rows 0..15  x tile 1
        v8f acc11 = {};   // rows 16..31 x tile 1

        for (int kk = 0; kk < HDIM; kk += 32) {
            v16bf a0 = load_frag_bf16(a0base + kk);
            v16bf a1 = load_frag_bf16(a1base + kk);

            v16bf b0, b1;
            if constexpr (PRECONV) {
                b0 = load_frag_bf16(wb0 + kk);
                b1 = load_frag_bf16(wb1 + kk);
                __builtin_prefetch(wb0 + kk + 512, 0, 0);   // stream W ahead
            } else {
                b0 = load_frag_f32(wf0 + kk);
                b1 = load_frag_f32(wf1 + kk);
                __builtin_prefetch(wf0 + kk + 256, 0, 0);
            }

            acc00 = __builtin_amdgcn_wmma_f32_16x16x32_bf16(
                false, a0, false, b0, (short)0, acc00, false, false);
            acc01 = __builtin_amdgcn_wmma_f32_16x16x32_bf16(
                false, a1, false, b0, (short)0, acc01, false, false);
            acc10 = __builtin_amdgcn_wmma_f32_16x16x32_bf16(
                false, a0, false, b1, (short)0, acc10, false, false);
            acc11 = __builtin_amdgcn_wmma_f32_16x16x32_bf16(
                false, a1, false, b1, (short)0, acc11, false, false);
        }

        // online log-sum-exp update + label logit capture (both tiles at once)
#pragma unroll
        for (int r = 0; r < 8; ++r) {
            float ca = acc00[r], cb = acc10[r];
            float nm = fmaxf(rm0[r], fmaxf(ca, cb));
            rs0[r] = rs0[r] * fexp2(L2E * (rm0[r] - nm))
                   + fexp2(L2E * (ca - nm)) + fexp2(L2E * (cb - nm));
            rm0[r] = nm;
            if (lab0[r] == vcol0) ll0[r] = ca;
            if (lab0[r] == vcol1) ll0[r] = cb;

            ca = acc01[r]; cb = acc11[r];
            nm = fmaxf(rm1[r], fmaxf(ca, cb));
            rs1[r] = rs1[r] * fexp2(L2E * (rm1[r] - nm))
                   + fexp2(L2E * (ca - nm)) + fexp2(L2E * (cb - nm));
            rm1[r] = nm;
            if (lab1[r] == vcol0) ll1[r] = ca;
            if (lab1[r] == vcol1) ll1[r] = cb;
        }
    }

    // ---- butterfly reduce across the 16 columns (stays within lane halves) ----
#pragma unroll
    for (int m = 1; m <= 8; m <<= 1) {
#pragma unroll
        for (int r = 0; r < 8; ++r) {
            float om = __shfl_xor(rm0[r], m, 32);
            float os = __shfl_xor(rs0[r], m, 32);
            float ol = __shfl_xor(ll0[r], m, 32);
            float nm = fmaxf(rm0[r], om);
            rs0[r] = rs0[r] * fexp2(L2E * (rm0[r] - nm)) + os * fexp2(L2E * (om - nm));
            rm0[r] = nm;
            ll0[r] = fmaxf(ll0[r], ol);

            om = __shfl_xor(rm1[r], m, 32);
            os = __shfl_xor(rs1[r], m, 32);
            ol = __shfl_xor(ll1[r], m, 32);
            nm = fmaxf(rm1[r], om);
            rs1[r] = rs1[r] * fexp2(L2E * (rm1[r] - nm)) + os * fexp2(L2E * (om - nm));
            rm1[r] = nm;
            ll1[r] = fmaxf(ll1[r], ol);
        }
    }

    // ---- cross-wave combine via LDS ----
    if (col == 0) {
#pragma unroll
        for (int r = 0; r < 8; ++r) {
            int row0 = khalf * 8 + r;
            float* p = red + (wave * MTILE + row0) * 3;
            p[0] = rm0[r]; p[1] = rs0[r]; p[2] = ll0[r];
            p = red + (wave * MTILE + row0 + 16) * 3;
            p[0] = rm1[r]; p[1] = rs1[r]; p[2] = ll1[r];
        }
    }
    __syncthreads();

    if (tid < MTILE) {
        float m = -INFINITY, s = 0.f, l = -INFINITY;
#pragma unroll
        for (int w = 0; w < NWAVES; ++w) {
            const float* p = red + (w * MTILE + tid) * 3;
            float om = p[0], os = p[1], ol = p[2];
            float nm = fmaxf(m, om);
            s = s * fexp2(L2E * (m - nm)) + os * fexp2(L2E * (om - nm));
            m = nm;
            l = fmaxf(l, ol);
        }
        int g = rowbase + tid;
        if (g < NTOK) {
            float* p = part + ((size_t)g * VSPLIT + vs) * 3;
            p[0] = m; p[1] = s; p[2] = l;
        }
    }
}

// One-shot f32 -> bf16 conversion of W into workspace (re-run every call).
__global__ __launch_bounds__(256)
void convert_w_kernel(const float* __restrict__ Wf, bf16_t* __restrict__ Wb)
{
    const size_t total = (size_t)VOCAB * HDIM;
    size_t i = ((size_t)blockIdx.x * blockDim.x + threadIdx.x) * 8;
    const size_t stride = (size_t)gridDim.x * blockDim.x * 8;
    for (; i < total; i += stride) {
        float4 a = ((const float4*)(Wf + i))[0];
        float4 b = ((const float4*)(Wf + i))[1];
        v8bf o;
        o[0]=(bf16_t)a.x; o[1]=(bf16_t)a.y; o[2]=(bf16_t)a.z; o[3]=(bf16_t)a.w;
        o[4]=(bf16_t)b.x; o[5]=(bf16_t)b.y; o[6]=(bf16_t)b.z; o[7]=(bf16_t)b.w;
        *(v8bf*)(Wb + i) = o;
    }
}

__global__ void finalize_kernel(const float* __restrict__ part,
                                const int*   __restrict__ labels,
                                float*       __restrict__ sums)
{
    int g = blockIdx.x * blockDim.x + threadIdx.x;
    if (g >= NTOK) return;
    float m = -INFINITY, s = 0.f, l = -INFINITY;
#pragma unroll
    for (int vs = 0; vs < VSPLIT; ++vs) {
        const float* p = part + ((size_t)g * VSPLIT + vs) * 3;
        float om = p[0], os = p[1], ol = p[2];
        float nm = fmaxf(m, om);
        s = s * fexp2(L2E * (m - nm)) + os * fexp2(L2E * (om - nm));
        m = nm;
        l = fmaxf(l, ol);
    }
    int lab = labels[g + 1];
    if (lab != IGNORE_INDEX) {
        float loss = m + flog2(s) * LN2 - l;   // -(logit_lab - logsumexp)
        atomicAdd(&sums[0], loss);
        atomicAdd(&sums[1], 1.0f);
    }
}

__global__ void init_kernel(float* sums)
{
    if (threadIdx.x < 2) sums[threadIdx.x] = 0.f;
}

__global__ void div_kernel(const float* __restrict__ sums, float* __restrict__ out)
{
    out[0] = sums[0] / sums[1];
}

extern "C" void kernel_launch(void* const* d_in, const int* in_sizes, int n_in,
                              void* d_out, int out_size, void* d_ws, size_t ws_size,
                              hipStream_t stream)
{
    const float* hs     = (const float*)d_in[0];   // hidden_states f32
    const int*   labels = (const int*)  d_in[1];   // labels (jax default int32)
    const float* Wf     = (const float*)d_in[2];   // weight f32 [V][H]
    // d_in[3] = mini_s (chunking hint) — irrelevant to the fused result.

    float* sums = (float*)d_ws;      // [0]=loss sum, [1]=count
    float* part = sums + 2;          // [NTOK][VSPLIT][3] partial lse state

    const size_t WB_OFF  = 256 * 1024;                   // past sums+part
    const size_t WB_SIZE = (size_t)VOCAB * HDIM * sizeof(bf16_t);
    bf16_t* Wb = (bf16_t*)((char*)d_ws + WB_OFF);
    const bool preconv = (ws_size >= WB_OFF + WB_SIZE);  // deterministic in ws_size

    float* out = (float*)d_out;

    init_kernel<<<1, 32, 0, stream>>>(sums);

    dim3 grid((NTOK + MTILE - 1) / MTILE, VSPLIT);
    size_t shmem = A_BYTES + (size_t)NWAVES * MTILE * 3 * sizeof(float);

    if (preconv) {
        const size_t total = (size_t)VOCAB * HDIM / 8;   // v8 chunks
        int cblocks = (int)((total + 255) / 256);
        convert_w_kernel<<<cblocks, 256, 0, stream>>>(Wf, Wb);
        fused_ce_kernel<true><<<grid, THREADS, shmem, stream>>>(hs, labels, Wf, Wb, part);
    } else {
        fused_ce_kernel<false><<<grid, THREADS, shmem, stream>>>(hs, labels, Wf, Wb, part);
    }

    finalize_kernel<<<(NTOK + 255) / 256, 256, 0, stream>>>(part, labels, sums);
    div_kernel<<<1, 1, 0, stream>>>(sums, out);
}
